// AttrsEncoderLayers_3418793968057
// MI455X (gfx1250) — matrix operand order repaired and checked by reading, not compiled, not run on hardware
//
#include <hip/hip_runtime.h>
#include <hip/hip_bf16.h>
#include <hip/hip_fp16.h>

// ---------------------------------------------------------------------------
// MI455X (gfx1250, wave32) implementation.
// GEMMs use v_wmma_f32_16x16x32_f16: M=16 edges per wave, N=128 (8 tiles),
// K=128 (4 steps). B matrices (128x128 f16, transposed to [n][k]) live in LDS.
// ---------------------------------------------------------------------------

typedef __attribute__((ext_vector_type(16))) _Float16 v16h;
typedef __attribute__((ext_vector_type(8)))  float    v8f;

#define WMMA16(a, b, c) \
  __builtin_amdgcn_wmma_f32_16x16x32_f16(false, (a), false, (b), (short)0, (c), false, false)

#define EPS       1e-5f
#define NEG_SLOPE 0.2f
#define NODES     50000
#define EDGES     200000

// ---------------- zero the stats region of the workspace -------------------
__global__ void zero_ws_kernel(float* p, int n) {
  for (int i = blockIdx.x * blockDim.x + threadIdx.x; i < n; i += gridDim.x * blockDim.x)
    p[i] = 0.0f;
}

// ---------------- generic column sum / sum-of-squares -----------------------
// x: [rows, C] row-major f32.  sum/sq: length C, accumulated with atomics.
__global__ void col_stats_kernel(const float* __restrict__ x, int rows, int C,
                                 float* __restrict__ sum, float* __restrict__ sq) {
  __shared__ float s1[128], s2[128];
  for (int i = threadIdx.x; i < C; i += blockDim.x) { s1[i] = 0.f; s2[i] = 0.f; }
  __syncthreads();
  int total = rows * C;
  for (int i = blockIdx.x * blockDim.x + threadIdx.x; i < total; i += gridDim.x * blockDim.x) {
    float v = x[i];
    int   c = i % C;
    atomicAdd(&s1[c], v);
    atomicAdd(&s2[c], v * v);
  }
  __syncthreads();
  for (int i = threadIdx.x; i < C; i += blockDim.x) {
    atomicAdd(&sum[i], s1[i]);
    atomicAdd(&sq[i],  s2[i]);
  }
}

// ---------------- prep1: fold BN0 into W1; transpose weights to f16 ---------
// W1s[k*128+c]  = a0[c] * W1[c*128+k]   (B for GEMM1, stored [n][k]-major)
// bias1[k]      = sum_c b0[c] * W1[c*128+k]
// Wgs[o*128+k]  = W_gat[k*128+o]        (B for GEMM2)
__global__ __launch_bounds__(128) void prep1_kernel(
    const float* __restrict__ sum0, const float* __restrict__ sq0,
    const float* __restrict__ gamma0, const float* __restrict__ beta0,
    const float* __restrict__ W1, const float* __restrict__ Wg,
    float* __restrict__ bias1, _Float16* __restrict__ W1s, _Float16* __restrict__ Wgs) {
  __shared__ float sb[128];
  int t = threadIdx.x;
  float cnt = (t < 64) ? (float)NODES : (float)EDGES;
  float m   = sum0[t] / cnt;
  float var = sq0[t] / cnt - m * m;
  float a   = gamma0[t] * rsqrtf(var + EPS);
  float b   = beta0[t] - m * a;
  sb[t] = b;
  __syncthreads();
  for (int k = 0; k < 128; ++k)
    W1s[k * 128 + t] = (_Float16)(a * W1[t * 128 + k]);
  float acc = 0.f;
  for (int c = 0; c < 128; ++c) acc += sb[c] * W1[c * 128 + t];
  bias1[t] = acc;
  for (int k = 0; k < 128; ++k)
    Wgs[t * 128 + k] = (_Float16)Wg[k * 128 + t];
}

// ---------------- prep2: BN1 scale/shift from y column stats ----------------
__global__ __launch_bounds__(128) void prep2_kernel(
    const float* __restrict__ ysum, const float* __restrict__ ysq,
    const float* __restrict__ gamma1, const float* __restrict__ beta1,
    float* __restrict__ a1, float* __restrict__ b1) {
  int t = threadIdx.x;
  float inv = 1.0f / (float)EDGES;
  float m   = ysum[t] * inv;
  float var = ysq[t] * inv - m * m;
  float a   = gamma1[t] * rsqrtf(var + EPS);
  a1[t] = a;
  b1[t] = beta1[t] - m * a;
}

// ---------------- GEMM1: y = h0_f16 @ W1' + bias1 (WMMA), + y col stats -----
// One wave handles 16 edge rows; block = 4 waves.
__global__ __launch_bounds__(128) void gemm1_kernel(
    const float* __restrict__ nodeA, const float* __restrict__ edgeA,
    const int* __restrict__ srcIdx, const _Float16* __restrict__ W1s,
    const float* __restrict__ bias1, _Float16* __restrict__ y,
    float* __restrict__ ysum, float* __restrict__ ysq) {
  __shared__ _Float16 sW[128 * 128];
  __shared__ float cs[128], cs2[128];
  {
    const uint4* s = (const uint4*)W1s;
    uint4* d = (uint4*)sW;
    for (int i = threadIdx.x; i < 2048; i += 128) d[i] = s[i];
  }
  if (threadIdx.x < 128) { cs[threadIdx.x] = 0.f; cs2[threadIdx.x] = 0.f; }
  __syncthreads();

  const int wid = threadIdx.x >> 5, lane = threadIdx.x & 31;
  const int l15 = lane & 15, hs = lane >> 4;
  const int gw = blockIdx.x * 4 + wid;
  const int rowBase = gw * 16;
  const int r = rowBase + l15;
  const int node = srcIdx[r];          // index_r gather

  v8f acc[8] = {};
  for (int kk = 0; kk < 4; ++kk) {
    v16h a;
    const int c0 = kk * 32 + hs * 16;  // contiguous k order: k = 16*half + e
#pragma unroll
    for (int e = 0; e < 16; ++e) {
      int c = c0 + e;
      float v = (c < 64) ? nodeA[node * 64 + c] : edgeA[r * 64 + (c - 64)];
      a[e] = (_Float16)v;
    }
#pragma unroll
    for (int t = 0; t < 8; ++t) {
      v16h b;
      const _Float16* bp = &sW[(t * 16 + l15) * 128 + kk * 32 + hs * 16];
#pragma unroll
      for (int e = 0; e < 16; ++e) b[e] = bp[e];
      acc[t] = WMMA16(a, b, acc[t]);
    }
  }
#pragma unroll
  for (int t = 0; t < 8; ++t) {
    const int n = t * 16 + l15;
    const float bn = bias1[n];
    float ps = 0.f, ps2 = 0.f;
#pragma unroll
    for (int v = 0; v < 8; ++v) {
      float val = acc[t][v] + bn;
      y[(size_t)(rowBase + hs * 8 + v) * 128 + n] = (_Float16)val;
      ps += val;
      ps2 += val * val;
    }
    atomicAdd(&cs[n], ps);
    atomicAdd(&cs2[n], ps2);
  }
  __syncthreads();
  if (threadIdx.x < 128) {
    atomicAdd(&ysum[threadIdx.x], cs[threadIdx.x]);
    atomicAdd(&ysq[threadIdx.x],  cs2[threadIdx.x]);
  }
}

// ---------------- GEMM2 + group attention + pooling (WMMA) ------------------
// xp = relu(BN1(y)) @ W_gat. 16 edges = 4 node-groups per wave; dense 4x4
// attention softmax per group; pooled h3[node] = sum_j w_j xp[j] + 4*bias.
__global__ __launch_bounds__(128) void gemm2_kernel(
    const _Float16* __restrict__ y, const _Float16* __restrict__ Wgs,
    const float* __restrict__ a1g, const float* __restrict__ b1g,
    const float* __restrict__ att_src, const float* __restrict__ att_dst,
    const float* __restrict__ bias_gat, const int* __restrict__ srcIdx,
    float* __restrict__ h3) {
  __shared__ _Float16 sW[128 * 128];
  __shared__ float sa1[128], sb1[128];
  __shared__ float sAs[4][16], sAd[4][16];
  {
    const uint4* s = (const uint4*)Wgs;
    uint4* d = (uint4*)sW;
    for (int i = threadIdx.x; i < 2048; i += 128) d[i] = s[i];
  }
  if (threadIdx.x < 128) { sa1[threadIdx.x] = a1g[threadIdx.x]; sb1[threadIdx.x] = b1g[threadIdx.x]; }
  __syncthreads();

  const int wid = threadIdx.x >> 5, lane = threadIdx.x & 31;
  const int l15 = lane & 15, hs = lane >> 4;
  const int gw = blockIdx.x * 4 + wid;
  const int rowBase = gw * 16;
  const int r = rowBase + l15;

  v8f acc[8] = {};
  for (int kk = 0; kk < 4; ++kk) {
    v16h a;
#pragma unroll
    for (int e = 0; e < 16; ++e) {
      int k = kk * 32 + hs * 16 + e;
      float v = (float)y[(size_t)r * 128 + k];
      v = sa1[k] * v + sb1[k];
      v = (v > 0.f) ? v : 0.f;                   // ReLU
      a[e] = (_Float16)v;
    }
#pragma unroll
    for (int t = 0; t < 8; ++t) {
      v16h b;
      const _Float16* bp = &sW[(t * 16 + l15) * 128 + kk * 32 + hs * 16];
#pragma unroll
      for (int e = 0; e < 16; ++e) b[e] = bp[e];
      acc[t] = WMMA16(a, b, acc[t]);
    }
  }

  // per-row attention logits a_s = xp.att_src, a_d = xp.att_dst
  float ps[8], pd[8];
#pragma unroll
  for (int v = 0; v < 8; ++v) { ps[v] = 0.f; pd[v] = 0.f; }
#pragma unroll
  for (int t = 0; t < 8; ++t) {
    const int n = t * 16 + l15;
    const float as_ = att_src[n], ad_ = att_dst[n];
#pragma unroll
    for (int v = 0; v < 8; ++v) { ps[v] += acc[t][v] * as_; pd[v] += acc[t][v] * ad_; }
  }
#pragma unroll
  for (int m = 1; m < 16; m <<= 1) {
#pragma unroll
    for (int v = 0; v < 8; ++v) {
      ps[v] += __shfl_xor(ps[v], m, 32);
      pd[v] += __shfl_xor(pd[v], m, 32);
    }
  }
  if (l15 == 0) {
#pragma unroll
    for (int v = 0; v < 8; ++v) { sAs[wid][hs * 8 + v] = ps[v]; sAd[wid][hs * 8 + v] = pd[v]; }
  }
  __syncthreads();

  // dense 4x4 softmax per group; w[j] = sum_i alpha_ij (this half owns 2 groups)
  float w[8];
#pragma unroll
  for (int g = 0; g < 2; ++g) {
    const int base = hs * 8 + g * 4;
    float wl0 = 0.f, wl1 = 0.f, wl2 = 0.f, wl3 = 0.f;
#pragma unroll
    for (int i = 0; i < 4; ++i) {
      const float di = sAd[wid][base + i];
      float e0 = sAs[wid][base + 0] + di; e0 = (e0 > 0.f) ? e0 : NEG_SLOPE * e0;
      float e1 = sAs[wid][base + 1] + di; e1 = (e1 > 0.f) ? e1 : NEG_SLOPE * e1;
      float e2 = sAs[wid][base + 2] + di; e2 = (e2 > 0.f) ? e2 : NEG_SLOPE * e2;
      float e3 = sAs[wid][base + 3] + di; e3 = (e3 > 0.f) ? e3 : NEG_SLOPE * e3;
      float mx = fmaxf(fmaxf(e0, e1), fmaxf(e2, e3));
      float x0 = __expf(e0 - mx), x1 = __expf(e1 - mx), x2 = __expf(e2 - mx), x3 = __expf(e3 - mx);
      float inv = 1.0f / (x0 + x1 + x2 + x3);
      wl0 += x0 * inv; wl1 += x1 * inv; wl2 += x2 * inv; wl3 += x3 * inv;
    }
    w[g * 4 + 0] = wl0; w[g * 4 + 1] = wl1; w[g * 4 + 2] = wl2; w[g * 4 + 3] = wl3;
  }

  // pooled per node (segment_sum over index_r; groups share one node id)
  const int nodeA = srcIdx[rowBase + hs * 8];
  const int nodeB = srcIdx[rowBase + hs * 8 + 4];
#pragma unroll
  for (int t = 0; t < 8; ++t) {
    const int n = t * 16 + l15;
    const float bg4 = 4.0f * bias_gat[n];
    float pA = 0.f, pB = 0.f;
#pragma unroll
    for (int v = 0; v < 4; ++v) pA += w[v] * acc[t][v];
#pragma unroll
    for (int v = 4; v < 8; ++v) pB += w[v] * acc[t][v];
    h3[(size_t)nodeA * 128 + n] = pA + bg4;
    h3[(size_t)nodeB * 128 + n] = pB + bg4;
  }
}

// ---------------- final BatchNorm over h3 -----------------------------------
__global__ void final_bn_kernel(const float* __restrict__ h3,
                                const float* __restrict__ hsum, const float* __restrict__ hsq,
                                const float* __restrict__ gamma_f, const float* __restrict__ beta_f,
                                float* __restrict__ out, int total) {
  const float inv = 1.0f / (float)NODES;
  for (int i = blockIdx.x * blockDim.x + threadIdx.x; i < total; i += gridDim.x * blockDim.x) {
    int c = i & 127;
    float m = hsum[c] * inv;
    float v = hsq[c] * inv - m * m;
    out[i] = (h3[i] - m) * rsqrtf(v + EPS) * gamma_f[c] + beta_f[c];
  }
}

// ---------------------------------------------------------------------------
extern "C" void kernel_launch(void* const* d_in, const int* in_sizes, int n_in,
                              void* d_out, int out_size, void* d_ws, size_t ws_size,
                              hipStream_t stream) {
  (void)n_in; (void)out_size; (void)ws_size;
  const float* node_attr = (const float*)d_in[0];
  const float* edge_attr = (const float*)d_in[1];
  const float* gamma0    = (const float*)d_in[2];
  const float* beta0     = (const float*)d_in[3];
  const float* W1        = (const float*)d_in[4];
  const float* gamma1    = (const float*)d_in[5];
  const float* beta1     = (const float*)d_in[6];
  const float* W_gat     = (const float*)d_in[7];
  const float* att_src   = (const float*)d_in[8];
  const float* att_dst   = (const float*)d_in[9];
  const float* bias_gat  = (const float*)d_in[10];
  const float* gamma_f   = (const float*)d_in[11];
  const float* beta_f    = (const float*)d_in[12];
  const int*   edge_idx  = (const int*)d_in[13];   // row 0 = index_r (src)

  const int N = in_sizes[0] / 64;   // 50000
  const int E = in_sizes[1] / 64;   // 200000

  // ---- workspace layout ----
  char* ws = (char*)d_ws;
  float* sum0  = (float*)ws;            // [128] BN0 col sums (node | edge)
  float* sq0   = sum0 + 128;            // [128]
  float* ysum  = sum0 + 256;            // [128] y col sums
  float* ysq   = sum0 + 384;            // [128]
  float* hsum  = sum0 + 512;            // [128] h3 col sums
  float* hsq   = sum0 + 640;            // [128]
  float* bias1 = sum0 + 768;            // [128]
  float* a1    = sum0 + 896;            // [128]
  float* b1    = sum0 + 1024;           // [128]
  _Float16* W1s = (_Float16*)(ws + 8192);                     // [128*128] f16
  _Float16* Wgs = (_Float16*)(ws + 8192 + 32768);             // [128*128] f16
  _Float16* yst = (_Float16*)(ws + 73728);                    // [E*128]  f16
  float*    h3  = (float*)(ws + 73728 + (size_t)E * 128 * 2); // [N*128]  f32

  float* out = (float*)d_out;
  const int* srcIdx = edge_idx;  // edge_index[0]

  const int gemmBlocks = (E / 16) / 4;   // 1 wave per 16 edges, 4 waves/block

  zero_ws_kernel<<<1, 256, 0, stream>>>(sum0, 1152);
  col_stats_kernel<<<512, 256, 0, stream>>>(node_attr, N, 64, sum0, sq0);
  col_stats_kernel<<<1024, 256, 0, stream>>>(edge_attr, E, 64, sum0 + 64, sq0 + 64);
  prep1_kernel<<<1, 128, 0, stream>>>(sum0, sq0, gamma0, beta0, W1, W_gat, bias1, W1s, Wgs);
  gemm1_kernel<<<gemmBlocks, 128, 0, stream>>>(node_attr, edge_attr, srcIdx, W1s, bias1,
                                               yst, ysum, ysq);
  prep2_kernel<<<1, 128, 0, stream>>>(ysum, ysq, gamma1, beta1, a1, b1);
  gemm2_kernel<<<gemmBlocks, 128, 0, stream>>>(yst, Wgs, a1, b1, att_src, att_dst,
                                               bias_gat, srcIdx, h3);
  col_stats_kernel<<<1024, 256, 0, stream>>>(h3, N, 128, hsum, hsq);
  final_bn_kernel<<<4096, 256, 0, stream>>>(h3, hsum, hsq, gamma_f, beta_f, out, N * 128);
}
